// RSNN_1374389535093
// MI455X (gfx1250) — matrix operand order repaired and checked by reading, not compiled, not run on hardware
//
#include <hip/hip_runtime.h>

// ---------------- constants (problem sizes are fixed by the reference) -----
#define THRESH 1.0f
#define BETA   0.9f
#define LN_EPS 1e-5f

static constexpr int Bsz = 64;
static constexpr int Tsz = 1024;
static constexpr int Dsz = 256;
static constexpr int Hsz = 256;

// ---------------- WMMA vector types (wave32, gfx1250) ----------------------
typedef __attribute__((ext_vector_type(16))) __bf16       v16bf;
typedef __attribute__((ext_vector_type(8)))  float        v8f;
typedef __attribute__((ext_vector_type(8)))  unsigned int v8u;

// float -> bf16 bits, round-to-nearest-even
__device__ __forceinline__ unsigned int f2bf(float f) {
  unsigned int u = __builtin_bit_cast(unsigned int, f);
  return (u + 0x7FFFu + ((u >> 16) & 1u)) >> 16;
}

// A-fragment (16x32 bf16, MxK) from a row-major bf16 LDS row.
// Lane layout (ISA 7.12.2): M = lane%16, half = lane/16.
// Elements 0..7  = K0..K0+7, elements 8..15 = K0+16..K0+23, K0 = 8*half + 32*kt.
__device__ __forceinline__ v16bf load_afrag(const unsigned short* row, int K0) {
  const uint4* p = (const uint4*)(row + K0);   // 8 bf16 per uint4
  uint4 lo = p[0];                             // K0   .. K0+7
  uint4 hi = p[2];                             // K0+16.. K0+23
  v8u w;
  w[0] = lo.x; w[1] = lo.y; w[2] = lo.z; w[3] = lo.w;
  w[4] = hi.x; w[5] = hi.y; w[6] = hi.z; w[7] = hi.w;
  return __builtin_bit_cast(v16bf, w);
}

// B-fragment (32x16 bf16, KxN) built from fp32 values of the weight row for
// this lane's N column: elements 0..7 <-> K = 8*half + e, 8..15 <-> K = 16 + 8*half + (e-8) (+32*kt).
__device__ __forceinline__ v16bf pack_bfrag_f32(const float* __restrict__ src) {
  v8u w;
#pragma unroll
  for (int e = 0; e < 8; ++e)
    w[e] = f2bf(src[2 * e + (e >= 4 ? 8 : 0)]) | (f2bf(src[2 * e + 1 + (e >= 4 ? 8 : 0)]) << 16);
  return __builtin_bit_cast(v16bf, w);
}

// ===========================================================================
// Phase 1: inp = LayerNorm(x @ W^T + b) -> written to d_out (in-place for P2)
// 512 blocks x 512 threads (16 waves). Each block: 128 rows (8 M-tiles of 16).
// Wave j owns output columns [16j, 16j+16); its W B-fragments live in VGPRs.
// x is stream-once: nontemporal loads keep L2 free for the inp buffer that
// phase 2 re-reads; next tile is prefetched (global_prefetch_b8).
// ===========================================================================
__global__ __launch_bounds__(512) void rsnn_proj_ln(
    const float* __restrict__ x, const float* __restrict__ W,
    const float* __restrict__ bias, const float* __restrict__ gamma,
    const float* __restrict__ beta, float* __restrict__ out)
{
  __shared__ __align__(16) unsigned short sA[16][256];  // x tile, bf16 (8 KB)
  __shared__ __align__(16) float sOut[16][256];         // pre-LN acc (16 KB)

  const int tid  = threadIdx.x;
  const int lane = tid & 31;
  const int wid  = tid >> 5;        // wave id = N-tile id (0..15)
  const int nl   = lane & 15;
  const int half = lane >> 4;
  const int ng   = wid * 16 + nl;   // global output column / W row

  // Preload W B-fragments for all 8 k-tiles (fp32 -> bf16), kept in VGPRs.
  v16bf wfr[8];
#pragma unroll
  for (int kt = 0; kt < 8; ++kt)
    wfr[kt] = pack_bfrag_f32(W + (size_t)ng * Dsz + kt * 32 + half * 8);

  const float bcol = bias[ng];
  const int row0blk = blockIdx.x * 128;

  for (int mt = 0; mt < 8; ++mt) {
    const int row0 = row0blk + mt * 16;
    __syncthreads();  // protect sA/sOut reuse across M-tiles

    // Stage 16x256 fp32 x-tile into LDS as bf16 (row-major), NT loads.
#pragma unroll
    for (int i = 0; i < 8; ++i) {
      int idx = tid + 512 * i;            // 0..4095
      int r = idx >> 8, c = idx & 255;
      float xv = __builtin_nontemporal_load(&x[(size_t)(row0 + r) * Dsz + c]);
      sA[r][c] = (unsigned short)f2bf(xv);
    }
    // Prefetch next M-tile (16 KB across 512 threads, 32 B each).
    if (mt < 7)
      __builtin_prefetch(&x[(size_t)(row0 + 16) * Dsz + tid * 8], 0, 3);
    __syncthreads();

    v8f acc = {};
#pragma unroll
    for (int kt = 0; kt < 8; ++kt) {
      v16bf a = load_afrag(&sA[nl][0], half * 8 + kt * 32);
      acc = __builtin_amdgcn_wmma_f32_16x16x32_bf16(
          false, a, false, wfr[kt], (short)0, acc, false, false);
    }

    // Scatter accumulator (D layout: M = i + 8*half, N = ng) + bias into LDS.
#pragma unroll
    for (int i = 0; i < 8; ++i)
      sOut[i + 8 * half][ng] = acc[i] + bcol;
    __syncthreads();

    // LayerNorm: wave `wid` reduces row `wid` (256 values, 8 per lane).
    float s = 0.f, s2 = 0.f, v[8];
#pragma unroll
    for (int i = 0; i < 8; ++i) {
      v[i] = sOut[wid][lane + 32 * i];
      s += v[i]; s2 += v[i] * v[i];
    }
#pragma unroll
    for (int off = 16; off > 0; off >>= 1) {
      s  += __shfl_xor(s,  off, 32);
      s2 += __shfl_xor(s2, off, 32);
    }
    const float mu  = s * (1.f / Hsz);
    const float var = s2 * (1.f / Hsz) - mu * mu;
    const float inv = rsqrtf(var + LN_EPS);
    float* orow = out + (size_t)(row0 + wid) * Hsz;
#pragma unroll
    for (int i = 0; i < 8; ++i) {
      int c = lane + 32 * i;
      orow[c] = (v[i] - mu) * inv * gamma[c] + beta[c];
    }
  }
}

// ===========================================================================
// Phase 2: sequential LIF scan, in place over d_out (inp -> mem).
// 4 blocks x 512 threads; block g owns batch rows [16g, 16g+16).
// - V B-fragments pre-converted to bf16 and pinned in VGPRs (64 VGPRs/wave).
// - `mem` lives permanently in WMMA C/D accumulator layout.
// - Spike tile double-buffered in LDS -> ONE barrier per step.
// - inp_t+1 loads issued during step t (L2 latency hidden behind WMMA chain).
// ===========================================================================
__global__ __launch_bounds__(512) void rsnn_scan(
    const float* __restrict__ V, const float* __restrict__ Vb,
    float* __restrict__ mems)
{
  __shared__ __align__(16) unsigned short sSpk[2][16][256];  // 2 x 8 KB

  const int tid  = threadIdx.x;
  const int lane = tid & 31;
  const int wid  = tid >> 5;
  const int nl   = lane & 15;
  const int half = lane >> 4;
  const int ng   = wid * 16 + nl;       // hidden column
  const int b0   = blockIdx.x * 16;     // first batch row of this block

  v16bf vfr[8];
#pragma unroll
  for (int kt = 0; kt < 8; ++kt)
    vfr[kt] = pack_bfrag_f32(V + (size_t)ng * Hsz + kt * 32 + half * 8);
  const float vb = Vb[ng];

  // spk(0) = 0 in buffer 0.
  {
    unsigned int* z = (unsigned int*)&sSpk[0][0][0];  // 2048 uints
#pragma unroll
    for (int i = 0; i < 4; ++i) z[tid + 512 * i] = 0u;
  }

  // Per-lane element addresses for t=0 (advance by Hsz per step).
  const size_t rowstride = (size_t)Tsz * Hsz;
  size_t off[8];
#pragma unroll
  for (int i = 0; i < 8; ++i)
    off[i] = (size_t)(b0 + i + 8 * half) * rowstride + (size_t)ng;

  // Preload inp(t=0).
  float ip[8];
#pragma unroll
  for (int i = 0; i < 8; ++i) ip[i] = mems[off[i]];

  v8f mem = {};

  for (int t = 0; t < Tsz; ++t) {
    __syncthreads();  // spk writes of step t-1 (to buf t&1) now visible

    // A fragments from the current spike buffer.
    v16bf afr[8];
    const unsigned short* srow = &sSpk[t & 1][nl][0];
#pragma unroll
    for (int kt = 0; kt < 8; ++kt)
      afr[kt] = load_afrag(srow, half * 8 + kt * 32);

    // C operand = beta*mem + inp_t + Vb - reset (reset == previous spike,
    // recomputed locally from the mem register: no LDS traffic).
    v8f c;
#pragma unroll
    for (int i = 0; i < 8; ++i) {
      const float reset = (mem[i] > THRESH) ? THRESH : 0.f;
      c[i] = BETA * mem[i] + ip[i] + vb - reset;
    }

    // Pipeline: issue inp(t+1) loads now; latency hidden by the WMMA chain.
    const int tn = (t + 1 < Tsz) ? t + 1 : t;  // dead value at t = Tsz-1
#pragma unroll
    for (int i = 0; i < 8; ++i)
      ip[i] = mems[off[i] + (size_t)tn * Hsz];

    v8f acc = c;
#pragma unroll
    for (int kt = 0; kt < 8; ++kt)
      acc = __builtin_amdgcn_wmma_f32_16x16x32_bf16(
          false, afr[kt], false, vfr[kt], (short)0, acc, false, false);
    mem = acc;

    // Write mems(t) and spikes for step t+1 into the other buffer.
#pragma unroll
    for (int i = 0; i < 8; ++i) {
      const int m = i + 8 * half;
      mems[off[i] + (size_t)t * Hsz] = mem[i];
      sSpk[(t + 1) & 1][m][ng] = (mem[i] > THRESH) ? (unsigned short)0x3F80
                                                   : (unsigned short)0x0000;
    }
  }
}

// ===========================================================================
extern "C" void kernel_launch(void* const* d_in, const int* in_sizes, int n_in,
                              void* d_out, int out_size, void* d_ws, size_t ws_size,
                              hipStream_t stream) {
  (void)in_sizes; (void)n_in; (void)out_size; (void)d_ws; (void)ws_size;
  const float* x     = (const float*)d_in[0];
  const float* W     = (const float*)d_in[1];
  const float* b     = (const float*)d_in[2];
  const float* gamma = (const float*)d_in[3];
  const float* beta  = (const float*)d_in[4];
  const float* V     = (const float*)d_in[5];
  const float* Vb    = (const float*)d_in[6];
  float* out = (float*)d_out;

  // Phase 1: input projection + LayerNorm, written into d_out (in-place input
  // for phase 2). 65536 rows / 128 rows per block = 512 blocks.
  rsnn_proj_ln<<<512, 512, 0, stream>>>(x, W, b, gamma, beta, out);

  // Phase 2: serial scan over T; batch-parallel (4 blocks x 16 rows).
  rsnn_scan<<<4, 512, 0, stream>>>(V, Vb, out);
}
